// Encoder1_83777632076446
// MI455X (gfx1250) — matrix operand order repaired and checked by reading, not compiled, not run on hardware
//
// DGCNN-VAE encoder for MI455X (gfx1250), wave32 + v_wmma_f32_16x16x32_f16.
//
// Design notes (compile-only; reasoning from MI455X specs):
//  * Pipeline is memory-bound (~80 GFLOP vs ~1.5 GB traffic @ 23.3 TB/s), so
//    activations are stored f16 [sample, channel] (halves HBM traffic, feeds
//    WMMA B-fragments as contiguous 32B loads), accumulation is fp32 in WMMA.
//  * Every GEMM wave computes a 64x16 output strip (4 WMMA accumulators):
//    the B fragment is loaded once per K-block and reused by 4 WMMAs
//    (4x less activation/gather traffic, 4x WMMA density per loop).
//  * Edge features (nbr-center || center) are gathered on the fly inside the
//    WMMA B-fragment builder -> the [S,2C] edge tensor (<=168MB) never exists.
//  * Distance Gram matrices are computed per-batch with a WMMA tile kernel
//    (A = query rows, reused across 4 neighbor tiles), then row-wise top-k.
//  * BN (training mode) = fp32 two-pass per-channel reduction over samples.
//  * Input order assumes JAX pytree flatten (sorted dict keys) of
//    {'eps','params','x'}; index map documented at kernel_launch.
//  * Workspace use: ~263 MB, laid out below (all regions 256B aligned).

#include <hip/hip_runtime.h>

typedef __attribute__((ext_vector_type(16))) _Float16 v16h;
typedef __attribute__((ext_vector_type(8)))  float    v8f;

#define NPTS 2048
#define NBATCH 8
#define SLOPE 0.2f
#define EPS_BN 1e-5f
#define TPB 256

#define WMMA_F16(a, b, c) \
  __builtin_amdgcn_wmma_f32_16x16x32_f16(false, (a), false, (b), (short)0, (c), false, false)

// ---------------------------------------------------------------- fragments

// A fragment 16x32 f16, contiguous k order (same as B): lane L -> row M=L%16, half hl=L/16;
// elems 0..15 -> K = kbase+hl*16+0..15.
__device__ __forceinline__ v16h frag_a_load(const _Float16* __restrict__ W,
                                            int lda, int tile_m, int kbase, int lane) {
  const int m  = tile_m + (lane & 15);
  const int hl = (lane >> 4) & 1;
  const _Float16* r0 = W + (size_t)m * lda + kbase + hl * 16;
  const _Float16* r1 = r0 + 8;
  v16h a;
#pragma unroll
  for (int e = 0; e < 8; ++e) a[e] = r0[e];
#pragma unroll
  for (int e = 0; e < 8; ++e) a[8 + e] = r1[e];
  return a;
}

// B-matrix 32x16 f16 layout: lane L -> col N=L%16; K = kbase + (L/16)*16 + e.
// Sample-major [S, ldb] activations make this one aligned 32-byte load.
__device__ __forceinline__ v16h frag_b_load_nc(const _Float16* __restrict__ X,
                                               long long s, int ldb, int kbase, int lane) {
  return *(const v16h*)(X + (size_t)s * ldb + kbase + ((lane >> 4) & 1) * 16);
}

__device__ __forceinline__ v16h frag_b_load(const _Float16* __restrict__ X,
                                            long long s, int ldb, int kbase,
                                            int lane, long long S) {
  if (s < S) return frag_b_load_nc(X, s, ldb, kbase, lane);
  v16h z = {};
  return z;
}

__device__ __forceinline__ float lrelu_f(float v) { return v > 0.f ? v : SLOPE * v; }

// C/D tile: lane L -> col N=L%16, rows M = tile_m + (L/16)*8 + r  (r=0..7).
__device__ __forceinline__ void store_tile(void* Y, int ldo, long long tile_s,
                                           int tile_m, int lane, long long S,
                                           const v8f& acc, const float* bias,
                                           int f32out) {
  long long s = tile_s + (lane & 15);
  if (s >= S) return;
  const int m0 = tile_m + ((lane >> 4) & 1) * 8;
  if (f32out) {
    float* o = (float*)Y + (size_t)s * ldo + m0;
#pragma unroll
    for (int r = 0; r < 8; ++r) o[r] = acc[r] + (bias ? bias[m0 + r] : 0.f);
  } else {
    _Float16* o = (_Float16*)Y + (size_t)s * ldo + m0;
#pragma unroll
    for (int r = 0; r < 8; ++r) o[r] = (_Float16)(acc[r] + (bias ? bias[m0 + r] : 0.f));
  }
}

// ---------------------------------------------------------------- GEMM kernels

// Y[s,o] = sum_k W[o,k] * X[s,k] (+bias). W f16 [O,lda], X f16 [S,ldb].
// One wave per 64x16 output strip (4 accumulators reuse one B fragment);
// 8 waves / block. Requires O % 64 == 0 (true for all layers here).
__global__ void k_gemm_tile(const _Float16* __restrict__ W, int lda,
                            const _Float16* __restrict__ X, int ldb,
                            void* __restrict__ Y, int ldo,
                            long long S, int kblocks,
                            const float* __restrict__ bias, int f32out) {
  const int lane = threadIdx.x & 31;
  const int wv   = threadIdx.x >> 5;
  const long long tile_s = ((long long)blockIdx.x * (blockDim.x >> 5) + wv) * 16;
  if (tile_s >= S) return;
  const int m0 = blockIdx.y * 64;
  v8f acc0 = {}, acc1 = {}, acc2 = {}, acc3 = {};
  for (int kb = 0; kb < kblocks; ++kb) {
    const int kbase = kb * 32;
    v16h b  = frag_b_load(X, tile_s + (lane & 15), ldb, kbase, lane, S);
    v16h a0 = frag_a_load(W, lda, m0,      kbase, lane);
    v16h a1 = frag_a_load(W, lda, m0 + 16, kbase, lane);
    v16h a2 = frag_a_load(W, lda, m0 + 32, kbase, lane);
    v16h a3 = frag_a_load(W, lda, m0 + 48, kbase, lane);
    acc0 = WMMA_F16(a0, b, acc0);
    acc1 = WMMA_F16(a1, b, acc1);
    acc2 = WMMA_F16(a2, b, acc2);
    acc3 = WMMA_F16(a3, b, acc3);
  }
  store_tile(Y, ldo, tile_s, m0,      lane, S, acc0, bias, f32out);
  store_tile(Y, ldo, tile_s, m0 + 16, lane, S, acc1, bias, f32out);
  store_tile(Y, ldo, tile_s, m0 + 32, lane, S, acc2, bias, f32out);
  store_tile(Y, ldo, tile_s, m0 + 48, lane, S, acc3, bias, f32out);
}

// Edge-conv layer 1 (O == 64): B-fragment built by gathering
// [nbr - center ; center] from xt [B*N, ldx] using knn idx [B*N*kk].
// Sample s = (b*N+n)*kk + j. S is always a multiple of 16 -> no lane guard;
// the expensive gathered fragment is reused by all 4 output-channel tiles.
__global__ void k_gemm_edge(const _Float16* __restrict__ W, int lda,
                            const _Float16* __restrict__ xt, int ldx, int C,
                            const int* __restrict__ idx, int kk,
                            _Float16* __restrict__ Y, int ldo,
                            int S, int kblocks) {
  const int lane = threadIdx.x & 31;
  const int wv   = threadIdx.x >> 5;
  const int tile_s = (blockIdx.x * (blockDim.x >> 5) + wv) * 16;
  if (tile_s >= S) return;
  const int hl = (lane >> 4) & 1;
  const int s  = tile_s + (lane & 15);

  const int p   = s / kk;                 // = b*NPTS + n
  const int nbr = idx[s];                 // idx[p*kk + j] == idx[s]
  const int b   = p / NPTS;
  const _Float16* pc = xt + (size_t)p * ldx;
  const _Float16* pn = xt + ((size_t)b * NPTS + nbr) * (size_t)ldx;

  v8f acc0 = {}, acc1 = {}, acc2 = {}, acc3 = {};
  for (int kb = 0; kb < kblocks; ++kb) {
    const int kbase = kb * 32;
    v16h bf;
    const int k0 = kbase + hl * 16;
#pragma unroll
    for (int e = 0; e < 16; ++e) {
      const int kx = k0 + e;
      float v;
      if (kx < C)          v = (float)pn[kx] - (float)pc[kx];
      else if (kx < 2 * C) v = (float)pc[kx - C];
      else                 v = 0.f;
      bf[e] = (_Float16)v;
    }
    v16h a0 = frag_a_load(W, lda, 0,  kbase, lane);
    v16h a1 = frag_a_load(W, lda, 16, kbase, lane);
    v16h a2 = frag_a_load(W, lda, 32, kbase, lane);
    v16h a3 = frag_a_load(W, lda, 48, kbase, lane);
    acc0 = WMMA_F16(a0, bf, acc0);
    acc1 = WMMA_F16(a1, bf, acc1);
    acc2 = WMMA_F16(a2, bf, acc2);
    acc3 = WMMA_F16(a3, bf, acc3);
  }
  _Float16* o = Y + (size_t)s * ldo + hl * 8;
#pragma unroll
  for (int r = 0; r < 8; ++r) o[r]      = (_Float16)acc0[r];
#pragma unroll
  for (int r = 0; r < 8; ++r) o[16 + r] = (_Float16)acc1[r];
#pragma unroll
  for (int r = 0; r < 8; ++r) o[32 + r] = (_Float16)acc2[r];
#pragma unroll
  for (int r = 0; r < 8; ++r) o[48 + r] = (_Float16)acc3[r];
}

// neg_dist[q,c] = 2*<x_q,x_c> - ||x_q||^2 - ||x_c||^2 for one batch.
// Each wave: 16 queries x 64 neighbors; the query (A) fragment is loaded
// once per K-block and reused across 4 neighbor tiles.
__global__ void k_dist(const _Float16* __restrict__ xt, int ldx,
                       const float* __restrict__ xx,
                       float* __restrict__ dist, int kblocks) {
  const int lane  = threadIdx.x & 31;
  const int wv    = threadIdx.x >> 5;
  const int qtile = (blockIdx.x * (blockDim.x >> 5) + wv) * 16;
  if (qtile >= NPTS) return;
  const int c0 = blockIdx.y * 64;
  v8f acc0 = {}, acc1 = {}, acc2 = {}, acc3 = {};
  for (int kb = 0; kb < kblocks; ++kb) {
    const int kbase = kb * 32;
    v16h a  = frag_a_load(xt, ldx, qtile, kbase, lane);
    v16h b0 = frag_b_load_nc(xt, c0      + (lane & 15), ldx, kbase, lane);
    v16h b1 = frag_b_load_nc(xt, c0 + 16 + (lane & 15), ldx, kbase, lane);
    v16h b2 = frag_b_load_nc(xt, c0 + 32 + (lane & 15), ldx, kbase, lane);
    v16h b3 = frag_b_load_nc(xt, c0 + 48 + (lane & 15), ldx, kbase, lane);
    acc0 = WMMA_F16(a, b0, acc0);
    acc1 = WMMA_F16(a, b1, acc1);
    acc2 = WMMA_F16(a, b2, acc2);
    acc3 = WMMA_F16(a, b3, acc3);
  }
  const int q0 = qtile + ((lane >> 4) & 1) * 8;
#pragma unroll
  for (int t = 0; t < 4; ++t) {
    const v8f& acc = (t == 0) ? acc0 : (t == 1) ? acc1 : (t == 2) ? acc2 : acc3;
    const int c = c0 + t * 16 + (lane & 15);
    const float xc = xx[c];
#pragma unroll
    for (int r = 0; r < 8; ++r)
      dist[(size_t)(q0 + r) * NPTS + c] = 2.f * acc[r] - xx[q0 + r] - xc;
  }
}

// ---------------------------------------------------------------- top-k

// One block per row: copy row to LDS, iterative argmax (ties -> lowest index),
// matching jax.lax.top_k ordering.
__global__ void k_topk(const float* __restrict__ dist, int* __restrict__ idxo, int kk) {
  __shared__ float sv[NPTS];
  __shared__ float rv[TPB];
  __shared__ int   ri[TPB];
  const int row = blockIdx.x;
  const float* src = dist + (size_t)row * NPTS;
  for (int i = threadIdx.x; i < NPTS; i += blockDim.x) sv[i] = src[i];
  __syncthreads();
  for (int t = 0; t < kk; ++t) {
    float bv = -3.4e38f; int bi = 0;
    for (int i = threadIdx.x; i < NPTS; i += blockDim.x) {
      float v = sv[i];
      if (v > bv) { bv = v; bi = i; }      // ascending i -> lowest idx on ties
    }
    rv[threadIdx.x] = bv; ri[threadIdx.x] = bi;
    __syncthreads();
    for (int off = blockDim.x >> 1; off > 0; off >>= 1) {
      if (threadIdx.x < off) {
        float ov = rv[threadIdx.x + off]; int oi = ri[threadIdx.x + off];
        if (ov > rv[threadIdx.x] || (ov == rv[threadIdx.x] && oi < ri[threadIdx.x])) {
          rv[threadIdx.x] = ov; ri[threadIdx.x] = oi;
        }
      }
      __syncthreads();
    }
    if (threadIdx.x == 0) { idxo[(size_t)row * kk + t] = ri[0]; sv[ri[0]] = -3.4e38f; }
    __syncthreads();
  }
}

// ---------------------------------------------------------------- BN / misc

__global__ void k_zero(float* p, int n) {
  int i = blockIdx.x * blockDim.x + threadIdx.x;
  if (i < n) p[i] = 0.f;
}

__global__ void k_bnstats(const _Float16* __restrict__ y, int S, int C,
                          float* __restrict__ sum, float* __restrict__ sumsq) {
  __shared__ float ls[1024], lq[1024];
  for (int c = threadIdx.x; c < C; c += blockDim.x) { ls[c] = 0.f; lq[c] = 0.f; }
  __syncthreads();
  const int total = S * C;
  for (int i = blockIdx.x * blockDim.x + threadIdx.x; i < total;
       i += gridDim.x * blockDim.x) {
    const int c = i % C;                 // C is a power of two everywhere
    const float v = (float)y[i];
    atomicAdd(&ls[c], v);
    atomicAdd(&lq[c], v * v);
  }
  __syncthreads();
  for (int c = threadIdx.x; c < C; c += blockDim.x) {
    atomicAdd(&sum[c], ls[c]);
    atomicAdd(&sumsq[c], lq[c]);
  }
}

__global__ void k_bnapply(_Float16* __restrict__ y, int S, int C,
                          const float* __restrict__ sum, const float* __restrict__ sumsq,
                          const float* __restrict__ g, const float* __restrict__ b) {
  const int i = blockIdx.x * blockDim.x + threadIdx.x;
  if (i >= S * C) return;
  const int c = i % C;
  const float m  = sum[c] / (float)S;
  const float var = sumsq[c] / (float)S - m * m;  // biased, like jnp.var
  float v = (float)y[i];
  v = g[c] * (v - m) * rsqrtf(var + EPS_BN) + b[c];
  y[i] = (_Float16)lrelu_f(v);
}

// max over k neighbors: y [(B*N*k), C] -> out[p*ldo + coff + c]
__global__ void k_maxk(const _Float16* __restrict__ y, int kk, int C,
                       _Float16* __restrict__ out, int ldo, int coff) {
  const int i = blockIdx.x * blockDim.x + threadIdx.x;
  if (i >= NBATCH * NPTS * C) return;
  const int p = i / C, c = i % C;
  const _Float16* src = y + ((size_t)p * kk) * C + c;
  float m = -3.4e38f;
  for (int j = 0; j < kk; ++j) m = fmaxf(m, (float)src[(size_t)j * C]);
  out[(size_t)p * ldo + coff + c] = (_Float16)m;
}

// max over the N points per (batch, channel)
__global__ void k_maxpt(const _Float16* __restrict__ h, int ld, int C,
                        _Float16* __restrict__ out) {
  const int i = blockIdx.x * blockDim.x + threadIdx.x;
  if (i >= NBATCH * C) return;
  const int b = i / C, c = i % C;
  const _Float16* p = h + ((size_t)b * NPTS) * ld + c;
  float m = -3.4e38f;
  for (int n = 0; n < NPTS; ++n) m = fmaxf(m, (float)p[(size_t)n * ld]);
  out[i] = (_Float16)m;
}

// pack raw points (+ optional 3x3 transform) to f16 [B*N,32] (zero pad) + ||x||^2
__global__ void k_pack3(const float* __restrict__ x, const float* __restrict__ tmat,
                        _Float16* __restrict__ xt, float* __restrict__ xx) {
  const int i = blockIdx.x * blockDim.x + threadIdx.x;
  if (i >= NBATCH * NPTS) return;
  const int b = i / NPTS, n = i % NPTS;
  const float* xp = x + ((size_t)b * 3) * NPTS + n;
  float c0 = xp[0], c1 = xp[NPTS], c2 = xp[2 * NPTS];
  float y0 = c0, y1 = c1, y2 = c2;
  if (tmat) {
    const float* t = tmat + b * 9;      // t[c*3+d]: y_d = sum_c x_c * t[c,d]
    y0 = c0 * t[0] + c1 * t[3] + c2 * t[6];
    y1 = c0 * t[1] + c1 * t[4] + c2 * t[7];
    y2 = c0 * t[2] + c1 * t[5] + c2 * t[8];
  }
  _Float16* o = xt + (size_t)i * 32;
  o[0] = (_Float16)y0; o[1] = (_Float16)y1; o[2] = (_Float16)y2;
  for (int c = 3; c < 32; ++c) o[c] = (_Float16)0.f;
  xx[i] = y0 * y0 + y1 * y1 + y2 * y2;
}

__global__ void k_xx64(const _Float16* __restrict__ xt, int ldx, float* __restrict__ xx) {
  const int i = blockIdx.x * blockDim.x + threadIdx.x;
  if (i >= NBATCH * NPTS) return;
  const _Float16* p = xt + (size_t)i * ldx;
  float s = 0.f;
  for (int c = 0; c < 64; ++c) { float v = (float)p[c]; s += v * v; }
  xx[i] = s;
}

// fp32 weight [O,K] -> f16 [O,Kpad] zero padded
__global__ void k_w2h(const float* __restrict__ src, _Float16* __restrict__ dst,
                      int O, int K, int Kpad) {
  const int i = blockIdx.x * blockDim.x + threadIdx.x;
  if (i >= O * Kpad) return;
  const int o = i / Kpad, kp = i % Kpad;
  dst[i] = (kp < K) ? (_Float16)src[(size_t)o * K + kp] : (_Float16)0.f;
}

// cat2[s, 0:512] = broadcast global max; cat2[s, 512:704] = cat[s,:]
__global__ void k_cat2(const _Float16* __restrict__ gmax, const _Float16* __restrict__ cat,
                       _Float16* __restrict__ out) {
  const int i = blockIdx.x * blockDim.x + threadIdx.x;
  if (i >= NBATCH * NPTS * 704) return;
  const int s = i / 704, c = i % 704;
  const int b = s / NPTS;
  out[i] = (c < 512) ? gmax[b * 512 + c] : cat[(size_t)s * 192 + (c - 512)];
}

// pooled[b, 0:256] = max over points, pooled[b, 256:512] = mean over points
__global__ void k_poolmm(const _Float16* __restrict__ feats, _Float16* __restrict__ pooled) {
  const int i = blockIdx.x * blockDim.x + threadIdx.x;
  if (i >= NBATCH * 256) return;
  const int b = i / 256, c = i % 256;
  const _Float16* p = feats + ((size_t)b * NPTS) * 256 + c;
  float mx = -3.4e38f, sm = 0.f;
  for (int n = 0; n < NPTS; ++n) { float v = (float)p[(size_t)n * 256]; mx = fmaxf(mx, v); sm += v; }
  pooled[b * 512 + c] = (_Float16)mx;
  pooled[b * 512 + 256 + c] = (_Float16)(sm * (1.f / NPTS));
}

// t[b,:9] = h[b,:256] @ t_w^T + t_b
__global__ void k_tmat(const _Float16* __restrict__ h, const float* __restrict__ tw,
                       const float* __restrict__ tb, float* __restrict__ t) {
  const int i = blockIdx.x * blockDim.x + threadIdx.x;
  if (i >= NBATCH * 9) return;
  const int b = i / 9, d = i % 9;
  float s = tb[d];
  for (int c = 0; c < 256; ++c) s += (float)h[b * 256 + c] * tw[d * 256 + c];
  t[i] = s;
}

// z = eps * exp(0.5*logvar) + mu ; out layout: [z | mu | logvar]
__global__ void k_zfinal(const float* __restrict__ eps, float* __restrict__ out) {
  const int i = blockIdx.x * blockDim.x + threadIdx.x;
  if (i >= NBATCH * 256) return;
  const float mu = out[2048 + i];
  const float lv = out[4096 + i];
  out[i] = eps[i] * expf(0.5f * lv) + mu;
}

// ---------------------------------------------------------------- launcher

static inline unsigned cdiv(long long a, long long b) { return (unsigned)((a + b - 1) / b); }

extern "C" void kernel_launch(void* const* d_in, const int* in_sizes, int n_in,
                              void* d_out, int out_size, void* d_ws, size_t ws_size,
                              hipStream_t stream) {
  (void)in_sizes; (void)n_in; (void)out_size; (void)ws_size;
  // JAX pytree flatten order (sorted dict keys) of {'eps','params','x'}:
  //  0 eps | 1-6 ec1 (W,g,b)x2 | 7-12 ec2 | 13-15 ec3
  //  logvar: 16 b1 17 b2 18 be1 19 g1 20 w1 21 w2
  //  22 m1_b 23 m1_g 24 m1_w | 25 m2a_b 26 m2a_g 27 m2a_w
  //  28 m2b_b 29 m2b_g 30 m2b_w | 31 m2c_bias 32 m2c_w
  //  mean: 33 b1 34 b2 35 be1 36 g1 37 w1 38 w2
  //  tn: 39 c1_b 40 c1_g 41 c1_w 42 c2_b 43 c2_g 44 c2_w 45 c3_b 46 c3_g
  //      47 c3_w 48 fc1_b 49 fc1_g 50 fc1_w 51 fc2_b 52 fc2_g 53 fc2_w
  //      54 t_b 55 t_w | 56 x
#define F32IN(i) ((const float*)d_in[(i)])

  char* ws = (char*)d_ws;
  float*     dist  = (float*)(ws + 0);                 //  16,777,216 B (per batch, reused)
  int*       idx   = (int*)(ws + 16777216);            //   2,621,440
  _Float16*  xt16  = (_Float16*)(ws + 19398656);       //   1,048,576  [B*N,32]
  float*     xx    = (float*)(ws + 20447232);          //      65,536
  _Float16*  wbuf  = (_Float16*)(ws + 20512768);       //   1,441,792  (weight f16 scratch)
  float*     stats = (float*)(ws + 21954560);          //       8,192  (sum|sumsq, 1024 each)
  char*      smallr= ws + 21962752;                    //      65,536
  _Float16*  cat   = (_Float16*)(ws + 22028288);       //   6,291,456  [B*N,192]
  _Float16*  yA    = (_Float16*)(ws + 28319744);       //  83,886,080
  _Float16*  yB    = (_Float16*)(ws + 112205824);      //  83,886,080
  _Float16*  pA    = (_Float16*)(ws + 196091904);      //  33,554,432
  _Float16*  pB    = (_Float16*)(ws + 229646336);      //  33,554,432  (end 263,200,768)

  _Float16* gmax   = (_Float16*)(smallr + 0);      // [8,512]
  _Float16* pooled = (_Float16*)(smallr + 8192);   // [8,512]
  _Float16* fcA    = (_Float16*)(smallr + 16384);  // [8,1024]
  _Float16* fcB    = (_Float16*)(smallr + 32768);  // [8,512]
  _Float16* fcC    = (_Float16*)(smallr + 40960);  // [8,256]
  float*    tmat   = (float*)(smallr + 49152);     // [8,9]

  auto gemm = [&](const float* Wsrc, int O, int K, int Kpad,
                  const _Float16* X, int ldb, void* Y, int ldo,
                  long long S, const float* bias, int f32out) {
    k_w2h<<<cdiv((long long)O * Kpad, TPB), TPB, 0, stream>>>(Wsrc, wbuf, O, K, Kpad);
    long long stiles = (S + 15) / 16;
    dim3 g(cdiv(stiles, 8), (unsigned)(O / 64));
    k_gemm_tile<<<g, TPB, 0, stream>>>(wbuf, Kpad, X, ldb, Y, ldo, S, Kpad / 32, bias, f32out);
  };
  auto egemm = [&](const float* Wsrc, int C, int Kpad, const _Float16* xt, int ldx,
                   int kk, _Float16* Y) {
    const int O = 64, S = NBATCH * NPTS * kk;
    k_w2h<<<cdiv((long long)O * Kpad, TPB), TPB, 0, stream>>>(Wsrc, wbuf, O, 2 * C, Kpad);
    dim3 g(cdiv(S / 16, 8), 1);
    k_gemm_edge<<<g, TPB, 0, stream>>>(wbuf, Kpad, xt, ldx, C, idx, kk, Y, O, S, Kpad / 32);
  };
  auto bn = [&](_Float16* y, long long S, int C, const float* g, const float* b) {
    const long long total = S * C;
    k_zero<<<8, TPB, 0, stream>>>(stats, 2048);
    unsigned sb = cdiv(total, TPB); if (sb > 4096) sb = 4096;
    k_bnstats<<<sb, TPB, 0, stream>>>(y, (int)S, C, stats, stats + 1024);
    k_bnapply<<<cdiv(total, TPB), TPB, 0, stream>>>(y, (int)S, C, stats, stats + 1024, g, b);
  };
  auto knn = [&](const _Float16* xt, int ldx, int kblocks, int kk) {
    for (int b = 0; b < NBATCH; ++b) {
      k_dist<<<dim3(16, 32), TPB, 0, stream>>>(xt + (size_t)b * NPTS * ldx, ldx,
                                               xx + b * NPTS, dist, kblocks);
      k_topk<<<NPTS, TPB, 0, stream>>>(dist, idx + (size_t)b * NPTS * kk, kk);
    }
  };
  const unsigned gBN = cdiv(NBATCH * NPTS, TPB);

  // ---------------- transform_net ----------------
  k_pack3<<<gBN, TPB, 0, stream>>>(F32IN(56), nullptr, xt16, xx);
  knn(xt16, 32, 1, 20);
  egemm(F32IN(41), 3, 32, xt16, 32, 20, yA);                       // c1: 6->64
  bn(yA, 327680, 64, F32IN(40), F32IN(39));
  gemm(F32IN(44), 128, 64, 64, yA, 64, yB, 128, 327680, nullptr, 0); // c2: 64->128
  bn(yB, 327680, 128, F32IN(43), F32IN(42));
  k_maxk<<<cdiv(16384LL * 128, TPB), TPB, 0, stream>>>(yB, 20, 128, pA, 128, 0);
  gemm(F32IN(47), 1024, 128, 128, pA, 128, pB, 1024, 16384, nullptr, 0); // c3
  bn(pB, 16384, 1024, F32IN(46), F32IN(45));
  k_maxpt<<<cdiv(8 * 1024, TPB), TPB, 0, stream>>>(pB, 1024, 1024, fcA);
  gemm(F32IN(50), 512, 1024, 1024, fcA, 1024, fcB, 512, 8, nullptr, 0);  // fc1
  bn(fcB, 8, 512, F32IN(49), F32IN(48));
  gemm(F32IN(53), 256, 512, 512, fcB, 512, fcC, 256, 8, nullptr, 0);     // fc2
  bn(fcC, 8, 256, F32IN(52), F32IN(51));
  k_tmat<<<1, 128, 0, stream>>>(fcC, F32IN(55), F32IN(54), tmat);
  k_pack3<<<gBN, TPB, 0, stream>>>(F32IN(56), tmat, xt16, xx);     // apply 3x3

  // ---------------- edge conv 1 (C=3, k=20) ----------------
  knn(xt16, 32, 1, 20);
  egemm(F32IN(1), 3, 32, xt16, 32, 20, yA);
  bn(yA, 327680, 64, F32IN(2), F32IN(3));
  gemm(F32IN(4), 64, 64, 64, yA, 64, yB, 64, 327680, nullptr, 0);
  bn(yB, 327680, 64, F32IN(5), F32IN(6));
  k_maxk<<<cdiv(16384LL * 64, TPB), TPB, 0, stream>>>(yB, 20, 64, cat, 192, 0);
  k_xx64<<<gBN, TPB, 0, stream>>>(cat, 192, xx);

  // ---------------- edge conv 2 (C=64, k=30) ----------------
  knn(cat, 192, 2, 30);
  egemm(F32IN(7), 64, 128, cat, 192, 30, yA);
  bn(yA, 491520, 64, F32IN(8), F32IN(9));
  gemm(F32IN(10), 64, 64, 64, yA, 64, yB, 64, 491520, nullptr, 0);
  bn(yB, 491520, 64, F32IN(11), F32IN(12));
  k_maxk<<<cdiv(16384LL * 64, TPB), TPB, 0, stream>>>(yB, 30, 64, cat, 192, 64);
  k_xx64<<<gBN, TPB, 0, stream>>>(cat + 64, 192, xx);

  // ---------------- edge conv 3 (C=64, k=40) ----------------
  knn(cat + 64, 192, 2, 40);
  egemm(F32IN(13), 64, 128, cat + 64, 192, 40, yA);
  bn(yA, 655360, 64, F32IN(14), F32IN(15));
  k_maxk<<<cdiv(16384LL * 64, TPB), TPB, 0, stream>>>(yA, 40, 64, cat, 192, 128);

  // ---------------- global MLP ----------------
  gemm(F32IN(24), 512, 192, 192, cat, 192, pA, 512, 16384, nullptr, 0);  // m1
  bn(pA, 16384, 512, F32IN(23), F32IN(22));
  k_maxpt<<<cdiv(8 * 512, TPB), TPB, 0, stream>>>(pA, 512, 512, gmax);
  k_cat2<<<cdiv(16384LL * 704, TPB), TPB, 0, stream>>>(gmax, cat, yA);   // [S,704]
  gemm(F32IN(27), 1024, 704, 704, yA, 704, pB, 1024, 16384, nullptr, 0); // m2a
  bn(pB, 16384, 1024, F32IN(26), F32IN(25));
  gemm(F32IN(30), 256, 1024, 1024, pB, 1024, pA, 256, 16384, nullptr, 0); // m2b
  bn(pA, 16384, 256, F32IN(29), F32IN(28));
  gemm(F32IN(32), 256, 256, 256, pA, 256, pB, 256, 16384, F32IN(31), 0);  // m2c + bias
  k_poolmm<<<cdiv(8 * 256, TPB), TPB, 0, stream>>>(pB, pooled);

  // ---------------- heads + reparameterize ----------------
  float* out = (float*)d_out;
  gemm(F32IN(37), 512, 512, 512, pooled, 512, fcB, 512, 8, F32IN(33), 0); // mean fc1
  bn(fcB, 8, 512, F32IN(36), F32IN(35));
  gemm(F32IN(38), 256, 512, 512, fcB, 512, out + 2048, 256, 8, F32IN(34), 1); // -> mu
  gemm(F32IN(20), 512, 512, 512, pooled, 512, fcB, 512, 8, F32IN(16), 0); // logvar fc1
  bn(fcB, 8, 512, F32IN(19), F32IN(18));
  gemm(F32IN(21), 256, 512, 512, fcB, 512, out + 4096, 256, 8, F32IN(17), 1); // -> logvar
  k_zfinal<<<cdiv(2048, TPB), TPB, 0, stream>>>(F32IN(0), out);
#undef F32IN
}
